// MMAttention_44341242364374
// MI455X (gfx1250) — matrix operand order repaired and checked by reading, not compiled, not run on hardware
//
#include <hip/hip_runtime.h>

#define HEADS 8
#define DHEAD 64
#define NSEQ  2329
#define NPAD  2336          // 146 * 16
#define DIM   512
#define M3    1536
#define QK_SCALE 0.125f     // 64^-0.5
#define NJT   73            // NPAD / 32

typedef __attribute__((ext_vector_type(16))) __bf16 v16bf;
typedef __attribute__((ext_vector_type(8)))  __bf16 v8bf;
typedef __attribute__((ext_vector_type(8)))  float  v8f;
typedef __attribute__((ext_vector_type(4)))  float  v4f;

__device__ __forceinline__ __bf16 f2bf(float f) { return static_cast<__bf16>(f); }

// A-frag (16x32 bf16, MxK): lane row = lane%16, elements = two 8-wide K runs
__device__ __forceinline__ v16bf frag_a_bf16(const __bf16* rowp, int kbase, int half) {
    v8bf lo = *(const v8bf*)(rowp + kbase + 8 * half);
    v8bf hi = *(const v8bf*)(rowp + kbase + 16 + 8 * half);
    v16bf a;
#pragma unroll
    for (int i = 0; i < 8; ++i) { a[i] = lo[i]; a[8 + i] = hi[i]; }
    return a;
}

// B-frag (32x16 bf16, KxN): lane col = lane%16, elements = two 8-wide K runs
__device__ __forceinline__ v16bf frag_b_bf16(const __bf16* rowp, int kbase, int half) {
    v8bf lo = *(const v8bf*)(rowp + kbase + 8 * half);
    v8bf hi = *(const v8bf*)(rowp + kbase + 16 + 8 * half);
    v16bf b;
#pragma unroll
    for (int i = 0; i < 8; ++i) { b[i] = lo[i]; b[8 + i] = hi[i]; }
    return b;
}

// ---------------------------------------------------------------------------
// Kernel 0: fp32 -> bf16 pre-conversion, 8 elements/thread.
// Elements in [valid, total) are written as zero (x row padding).
// ---------------------------------------------------------------------------
__global__ __launch_bounds__(256) void cvt_bf16_kernel(
    const float* __restrict__ src, __bf16* __restrict__ dst,
    int total, int valid)
{
    const int base = (blockIdx.x * 256 + threadIdx.x) * 8;
    if (base >= total) return;
    v8bf o;
    if (base < valid) {
        v4f a = *(const v4f*)(src + base);
        v4f b = *(const v4f*)(src + base + 4);
#pragma unroll
        for (int i = 0; i < 4; ++i) { o[i] = f2bf(a[i]); o[4 + i] = f2bf(b[i]); }
    } else {
#pragma unroll
        for (int i = 0; i < 8; ++i) o[i] = f2bf(0.0f);
    }
    *(v8bf*)(dst + base) = o;
}

// ---------------------------------------------------------------------------
// Kernel 1: qkv = x @ w^T (bf16 WMMA, f32 accum). One wave per 16x64 tile,
// 4 accumulators share one A-frag. Software pipelined: chunk kc+1 operands
// are prefetched before chunk kc's 4 WMMAs issue, hiding load latency.
// ---------------------------------------------------------------------------
__global__ __launch_bounds__(32) void qkv_gemm_kernel(
    const __bf16* __restrict__ xb, const __bf16* __restrict__ wb,
    __bf16* __restrict__ qs, __bf16* __restrict__ kb,
    __bf16* __restrict__ vT, float* __restrict__ vf)
{
    const int lane  = threadIdx.x;
    const int half  = lane >> 4;
    const int lan16 = lane & 15;
    const int rt = blockIdx.x;              // 146 row tiles
    const int ct = blockIdx.y;              // 24 tiles of 64 cols

    const __bf16* xrow = xb + (size_t)(rt * 16 + lan16) * DIM;
    const __bf16* wrow = wb + (size_t)(ct * 64 + lan16) * DIM;

    v8f acc[4] = {{}, {}, {}, {}};
    v16bf a = frag_a_bf16(xrow, 0, half);
    v16bf b[4];
#pragma unroll
    for (int t = 0; t < 4; ++t) b[t] = frag_b_bf16(wrow + (size_t)t * 16 * DIM, 0, half);

#pragma unroll 3
    for (int kc = 0; kc < DIM / 32 - 1; ++kc) {
        const int kn = (kc + 1) * 32;
        // prefetch next chunk's operands (latency hidden under 4 WMMAs)
        v16bf an = frag_a_bf16(xrow, kn, half);
        v16bf bn[4];
#pragma unroll
        for (int t = 0; t < 4; ++t) bn[t] = frag_b_bf16(wrow + (size_t)t * 16 * DIM, kn, half);
#pragma unroll
        for (int t = 0; t < 4; ++t)
            acc[t] = __builtin_amdgcn_wmma_f32_16x16x32_bf16(
                false, a, false, b[t], (short)0, acc[t], false, false);
        a = an;
#pragma unroll
        for (int t = 0; t < 4; ++t) b[t] = bn[t];
    }
#pragma unroll
    for (int t = 0; t < 4; ++t)
        acc[t] = __builtin_amdgcn_wmma_f32_16x16x32_bf16(
            false, a, false, b[t], (short)0, acc[t], false, false);

    const int cg  = ct * 64;                // 64-col tile = one (seg, head)
    const int seg = cg / 512;               // 0=q 1=k 2=v
    const int h   = (cg % 512) / 64;
#pragma unroll
    for (int t = 0; t < 4; ++t) {
        const int d = t * 16 + lan16;
#pragma unroll
        for (int r = 0; r < 8; ++r) {
            const int rr = rt * 16 + r + 8 * half;      // D row mapping
            const float val = acc[t][r];
            if (seg == 0) {
                qs[((size_t)h * NPAD + rr) * 64 + d] = f2bf(val * QK_SCALE);
            } else if (seg == 1) {
                kb[((size_t)h * NPAD + rr) * 64 + d] = f2bf(val);
            } else {
                vf[((size_t)h * NPAD + rr) * 64 + d] = val;
                vT[((size_t)h * 64 + d) * NPAD + rr] = f2bf(val);
            }
        }
    }
}

// ---------------------------------------------------------------------------
// Kernel 2: flash attention. One wave per 16 query rows per head.
// Per 32-key chunk: 4 WMMAs (S) + online softmax + LDS transpose of P +
// 4 WMMAs (P·V). k-frags are double-buffered across chunks; V-frag loads
// issue before the softmax so their latency hides under VALU/exp work.
// ---------------------------------------------------------------------------
__global__ __launch_bounds__(32) void attn_kernel(
    const __bf16* __restrict__ qs, const __bf16* __restrict__ kb,
    const __bf16* __restrict__ vT, float* __restrict__ out)
{
    __shared__ __attribute__((aligned(16))) __bf16 p_lds[16 * 40]; // 40: bank-pad
    const int lane  = threadIdx.x;
    const int half  = lane >> 4;
    const int lan16 = lane & 15;
    const int rt    = blockIdx.x;            // 146 query tiles
    const int h     = blockIdx.y;            // 8 heads
    const int ibase = rt * 16;

    const __bf16* qrow = qs + ((size_t)h * NPAD + ibase + lan16) * 64;
    const v16bf qa0 = frag_a_bf16(qrow, 0, half);
    const v16bf qa1 = frag_a_bf16(qrow, 32, half);

    const __bf16* kbase_p = kb + ((size_t)h * NPAD + lan16) * 64;
    const __bf16* vbase_p = vT + ((size_t)h * 64 + lan16) * NPAD;

    v8f o0 = {}, o1 = {}, o2 = {}, o3 = {};
    float mrow[8], lrow[8];
#pragma unroll
    for (int r = 0; r < 8; ++r) { mrow[r] = -3.0e38f; lrow[r] = 0.0f; }

    // prime k-frags for chunk 0
    v16bf kf[4];
    kf[0] = frag_b_bf16(kbase_p,           0,  half);
    kf[1] = frag_b_bf16(kbase_p,           32, half);
    kf[2] = frag_b_bf16(kbase_p + 16 * 64, 0,  half);
    kf[3] = frag_b_bf16(kbase_p + 16 * 64, 32, half);

    for (int jt = 0; jt < NJT; ++jt) {
        const int jbase = jt * 32;
        // ---- issue V-frag loads early: consumed only after the softmax
        const __bf16* vb = vbase_p + jbase;
        v16bf vf0 = frag_b_bf16(vb,             0, half);
        v16bf vf1 = frag_b_bf16(vb + 16 * NPAD, 0, half);
        v16bf vf2 = frag_b_bf16(vb + 32 * NPAD, 0, half);
        v16bf vf3 = frag_b_bf16(vb + 48 * NPAD, 0, half);

        // ---- S = q kT : two 16x16 tiles (cols jbase..+15, jbase+16..+31)
        v8f s1 = {}, s2 = {};
        s1 = __builtin_amdgcn_wmma_f32_16x16x32_bf16(false, qa0, false, kf[0], (short)0, s1, false, false);
        s1 = __builtin_amdgcn_wmma_f32_16x16x32_bf16(false, qa1, false, kf[1], (short)0, s1, false, false);
        s2 = __builtin_amdgcn_wmma_f32_16x16x32_bf16(false, qa0, false, kf[2], (short)0, s2, false, false);
        s2 = __builtin_amdgcn_wmma_f32_16x16x32_bf16(false, qa1, false, kf[3], (short)0, s2, false, false);

        // ---- prefetch next chunk's k-frags (hidden under softmax + PV)
        if (jt + 1 < NJT) {
            const __bf16* kn = kbase_p + (size_t)(jbase + 32) * 64;
            kf[0] = frag_b_bf16(kn,           0,  half);
            kf[1] = frag_b_bf16(kn,           32, half);
            kf[2] = frag_b_bf16(kn + 16 * 64, 0,  half);
            kf[3] = frag_b_bf16(kn + 16 * 64, 32, half);
        }

        const bool bad0 = (jbase + lan16)      >= NSEQ;
        const bool bad1 = (jbase + 16 + lan16) >= NSEQ;
#pragma unroll
        for (int r = 0; r < 8; ++r) {
            float a = bad0 ? -1.0e30f : s1[r];
            float b = bad1 ? -1.0e30f : s2[r];
            // row max over 32 cols (16-lane group reduction, wave32)
            float mx = fmaxf(a, b);
            mx = fmaxf(mx, __shfl_xor(mx, 1, 32));
            mx = fmaxf(mx, __shfl_xor(mx, 2, 32));
            mx = fmaxf(mx, __shfl_xor(mx, 4, 32));
            mx = fmaxf(mx, __shfl_xor(mx, 8, 32));
            const float mn    = fmaxf(mrow[r], mx);
            const float alpha = __expf(mrow[r] - mn);
            const float p0 = __expf(a - mn);
            const float p1 = __expf(b - mn);
            float rs = p0 + p1;
            rs += __shfl_xor(rs, 1, 32);
            rs += __shfl_xor(rs, 2, 32);
            rs += __shfl_xor(rs, 4, 32);
            rs += __shfl_xor(rs, 8, 32);
            lrow[r] = lrow[r] * alpha + rs;
            mrow[r] = mn;
            o0[r] *= alpha; o1[r] *= alpha; o2[r] *= alpha; o3[r] *= alpha;
            // stage P (D-layout -> [row][col]) for the A-frag transpose
            p_lds[(r + 8 * half) * 40 + lan16]      = f2bf(p0);
            p_lds[(r + 8 * half) * 40 + 16 + lan16] = f2bf(p1);
        }
        asm volatile("" ::: "memory");   // keep LDS store->load order
        // ---- P·V : P A-frag (16x32) from LDS, V B-frags already in flight
        v16bf pa;
        {
            const __bf16* pr = p_lds + lan16 * 40;
            v8bf lo = *(const v8bf*)(pr + 8 * half);
            v8bf hi = *(const v8bf*)(pr + 16 + 8 * half);
#pragma unroll
            for (int i = 0; i < 8; ++i) { pa[i] = lo[i]; pa[8 + i] = hi[i]; }
        }
        o0 = __builtin_amdgcn_wmma_f32_16x16x32_bf16(false, pa, false, vf0, (short)0, o0, false, false);
        o1 = __builtin_amdgcn_wmma_f32_16x16x32_bf16(false, pa, false, vf1, (short)0, o1, false, false);
        o2 = __builtin_amdgcn_wmma_f32_16x16x32_bf16(false, pa, false, vf2, (short)0, o2, false, false);
        o3 = __builtin_amdgcn_wmma_f32_16x16x32_bf16(false, pa, false, vf3, (short)0, o3, false, false);
    }

#pragma unroll
    for (int r = 0; r < 8; ++r) {
        const int row = ibase + r + 8 * half;
        if (row < NSEQ) {
            const float inv = 1.0f / lrow[r];
            float* ob = out + (size_t)row * DIM + h * 64;
            ob[lan16]      = o0[r] * inv;
            ob[16 + lan16] = o1[r] * inv;
            ob[32 + lan16] = o2[r] * inv;
            ob[48 + lan16] = o3[r] * inv;
        }
    }
}

// ---------------------------------------------------------------------------
// Kernel 3: depthwise (33,1) conv over v along the sequence, added into out.
// ---------------------------------------------------------------------------
__global__ __launch_bounds__(256) void conv_res_kernel(
    const float* __restrict__ vf, const float* __restrict__ wres,
    float* __restrict__ out)
{
    const int idx = blockIdx.x * 256 + threadIdx.x;
    if (idx >= NSEQ * DIM) return;
    const int i = idx / DIM, c = idx % DIM;
    const int h = c / 64,   d = c % 64;
    const float* vb = vf + (size_t)h * NPAD * 64 + d;
    const float* wr = wres + h * 33;
    float s = 0.0f;
#pragma unroll
    for (int t = 0; t < 33; ++t) {
        const int j = i + t - 16;
        if (j >= 0 && j < NSEQ) s += wr[t] * vb[(size_t)j * 64];
    }
    out[idx] += s;
}

extern "C" void kernel_launch(void* const* d_in, const int* in_sizes, int n_in,
                              void* d_out, int out_size, void* d_ws, size_t ws_size,
                              hipStream_t stream) {
    const float* x     = (const float*)d_in[0];
    const float* w_qkv = (const float*)d_in[1];
    const float* w_res = (const float*)d_in[2];
    float* out = (float*)d_out;

    char* ws = (char*)d_ws;
    const size_t nElem = (size_t)HEADS * NPAD * DHEAD;   // 1,196,032
    __bf16* qs = (__bf16*)(ws);                          // q*scale bf16 [h][n][d]
    __bf16* kb = (__bf16*)(ws + nElem * 2);              // k       bf16 [h][n][d]
    __bf16* vT = (__bf16*)(ws + nElem * 4);              // v^T     bf16 [h][d][n]
    float*  vf = (float*) (ws + nElem * 6);              // v       f32  [h][n][d]
    __bf16* xb = (__bf16*)(ws + nElem * 10);             // x  bf16 [NPAD][512]
    __bf16* wb = (__bf16*)(ws + nElem * 10 + (size_t)NPAD * DIM * 2); // w bf16 [1536][512]
    (void)in_sizes; (void)n_in; (void)out_size; (void)ws_size;

    const int xTot = NPAD * DIM, xVal = NSEQ * DIM;
    const int wTot = M3 * DIM;
    cvt_bf16_kernel<<<(xTot / 8 + 255) / 256, 256, 0, stream>>>(x, xb, xTot, xVal);
    cvt_bf16_kernel<<<(wTot / 8 + 255) / 256, 256, 0, stream>>>(w_qkv, wb, wTot, wTot);
    qkv_gemm_kernel<<<dim3(NPAD / 16, M3 / 64), 32, 0, stream>>>(xb, wb, qs, kb, vT, vf);
    attn_kernel<<<dim3(NPAD / 16, HEADS), 32, 0, stream>>>(qs, kb, vT, out);
    conv_res_kernel<<<(NSEQ * DIM + 255) / 256, 256, 0, stream>>>(vf, w_res, out);
}
